// InteractionLogisticRegression_20169166422772
// MI455X (gfx1250) — matrix operand order repaired and checked by reading, not compiled, not run on hardware
//
#include <hip/hip_runtime.h>

// Problem constants (from reference)
#define B_SZ 2048
#define I_SZ 6156
#define O_SZ 3
#define IO_SZ (I_SZ + O_SZ)

// GEMM tiling: 256 threads = 8 wave32 in a 4x2 grid; each wave owns a 32x32
// macro-tile = 4 C tiles of 16x16 -> 12 WMMAs per K-step (split-bf16 x3).
#define BM 128
#define BN 64
#define BK 32
#define NKT_FULL 192   // tiles 0..191 are full (192*32 = 6144 <= 6156); tile 192 partial

typedef __bf16 bf16_t;
typedef __attribute__((ext_vector_type(16))) bf16_t v16bf;
typedef __attribute__((ext_vector_type(2)))  bf16_t v2bf;
typedef __attribute__((ext_vector_type(8)))  float  v8f;
typedef __attribute__((ext_vector_type(16))) unsigned short v16us;

#if defined(__has_builtin)
#if __has_builtin(__builtin_amdgcn_cvt_pk_bf16_f32)
#define HAVE_PK_BF16 1
#endif
#endif

__device__ __forceinline__ unsigned short f2bf(float f) {
  unsigned int u = __builtin_bit_cast(unsigned int, f);
  u += 0x7FFFu + ((u >> 16) & 1u);           // round-to-nearest-even
  return (unsigned short)(u >> 16);
}

// Split an fp32 pair into a packed bf16-hi pair (truncation) and a packed
// bf16-lo pair (RNE of the exact residual).  hi+lo ~ 16-bit mantissa accuracy.
__device__ __forceinline__ void split_pair(float f0, float f1,
                                           unsigned int& hi, unsigned int& lo) {
  unsigned int u0 = __builtin_bit_cast(unsigned int, f0);
  unsigned int u1 = __builtin_bit_cast(unsigned int, f1);
  hi = (u0 >> 16) | (u1 & 0xFFFF0000u);
  float h0 = __builtin_bit_cast(float, u0 & 0xFFFF0000u);
  float h1 = __builtin_bit_cast(float, u1 & 0xFFFF0000u);
#ifdef HAVE_PK_BF16
  v2bf pl = __builtin_amdgcn_cvt_pk_bf16_f32(f0 - h0, f1 - h1);
  lo = __builtin_bit_cast(unsigned int, pl);
#else
  unsigned int l0 = f2bf(f0 - h0);
  unsigned int l1 = f2bf(f1 - h1);
  lo = l0 | (l1 << 16);
#endif
}

#define WMMA_BF16(A, B, C) \
  __builtin_amdgcn_wmma_f32_16x16x32_bf16(false, (A), false, (B), (short)0, (C), false, false)

// ---------------------------------------------------------------------------
// Kernel 0: zero the [B,O] fp32 accumulator in workspace
// ---------------------------------------------------------------------------
__global__ void zero_acc_kernel(float* __restrict__ p, int n) {
  int i = blockIdx.x * 256 + threadIdx.x;
  if (i < n) p[i] = 0.0f;
}

// ---------------------------------------------------------------------------
// Kernel 1: T = x @ w_int.T via split-bf16 WMMA, fused with
//           interactions = x * T and projection onto fin_w[:, O:].
//           LDS tiles live in the exact ISA 7.12.2 fragment layout so
//           fragment loads are plain ds_load_b128 pairs.  Interior K tiles
//           run unpredicated with loop-carried pointers; ragged K/N edges
//           take uniform-branch guarded variants only where needed.
// ---------------------------------------------------------------------------
__global__ __launch_bounds__(256)
void interaction_gemm_kernel(const float* __restrict__ x,
                             const float* __restrict__ w_int,
                             const float* __restrict__ fin_w,
                             float* __restrict__ gacc) {
  // Fragment-layout LDS: [rowtile][lane][8 x u32(=bf16 pair)]
  __shared__ __align__(32) unsigned int sAh[8 * 32 * 8], sAl[8 * 32 * 8]; // 8 KB each
  __shared__ __align__(32) unsigned int sBh[4 * 32 * 8], sBl[4 * 32 * 8]; // 4 KB each
  __shared__ float sacc[BM * O_SZ];

  const int tid  = threadIdx.x;
  const int lane = tid & 31;
  const int wid  = tid >> 5;
  const int wm   = wid & 3;    // 4 row-wave groups (32 rows each)
  const int wn   = wid >> 2;   // 2 col-wave groups (32 cols each)
  const int lrow = lane & 15;
  const int hsel = lane >> 4;

  const int row0 = blockIdx.y * BM;   // batch rows (2048 % 128 == 0)
  const int col0 = blockIdx.x * BN;   // feature cols (edge zero-padded)
  const bool fullN = (col0 + BN) <= I_SZ;   // block-uniform N-edge flag

  for (int i = tid; i < BM * O_SZ; i += 256) sacc[i] = 0.0f;

  // Loop-carried per-thread tile pointers; advance one K-tile (128 B) per use.
  const float2* px[8];
  const float2* pw[4];
  bool          valW[4];
#pragma unroll
  for (int j = 0; j < 8; ++j) {
    int pairIdx = tid + 256 * j;            // BM*BK/2 = 2048 pairs
    int m = pairIdx >> 4, t = pairIdx & 15;
    px[j] = reinterpret_cast<const float2*>(x + ((size_t)(row0 + m) * I_SZ + 2 * t));
  }
#pragma unroll
  for (int j = 0; j < 4; ++j) {
    int pairIdx = tid + 256 * j;            // BN*BK/2 = 1024 pairs
    int n = pairIdx >> 4, t = pairIdx & 15;
    int gn = col0 + n;
    valW[j] = gn < I_SZ;
    pw[j] = reinterpret_cast<const float2*>(
        w_int + ((size_t)(valW[j] ? gn : 0) * I_SZ + 2 * t));
  }

  v8f cc[2][2] = {};

  // ---- load variants (each call consumes one K-tile and advances pointers) ----
  auto loadX_fast = [&](float2* r) {         // all 32 k valid
#pragma unroll
    for (int j = 0; j < 8; ++j) { r[j] = *px[j]; px[j] += BK / 2; }
  };
  auto loadX_safe = [&](float2* r, int kt) { // K edge / past-the-end
#pragma unroll
    for (int j = 0; j < 8; ++j) {
      int t = (tid + 256 * j) & 15;
      int gk = kt * BK + 2 * t;              // even; gk<I => gk+1<I (I even)
      float2 v = make_float2(0.0f, 0.0f);
      if (gk < I_SZ) v = *px[j];
      px[j] += BK / 2;
      r[j] = v;
    }
  };
  auto loadW_fast = [&](float2* r) {         // interior block, full K
#pragma unroll
    for (int j = 0; j < 4; ++j) { r[j] = *pw[j]; pw[j] += BK / 2; }
  };
  auto loadW_safe = [&](float2* r, int kt) { // N and/or K edge
#pragma unroll
    for (int j = 0; j < 4; ++j) {
      int t = (tid + 256 * j) & 15;
      int gk = kt * BK + 2 * t;
      float2 v = make_float2(0.0f, 0.0f);
      if (valW[j] && gk < I_SZ) v = *pw[j];
      pw[j] += BK / 2;
      r[j] = v;
    }
  };

  // Convert fp32 pairs -> packed bf16 hi/lo, scatter into fragment layout.
  auto storeT = [&](const float2* rx, const float2* rw) {
#pragma unroll
    for (int j = 0; j < 8; ++j) {
      int pairIdx = tid + 256 * j;
      int m = pairIdx >> 4, k = (pairIdx & 15) * 2;
      unsigned int hi, lo;
      split_pair(rx[j].x, rx[j].y, hi, lo);
      // A layout: lane = (m%16) + 16*((k>>3)&1); pair slot interleaves k blocks
      int slot = (((m >> 4) * 32 + (m & 15) + 16 * ((k >> 3) & 1)) << 3)
               + ((k & 7) >> 1) + ((k & 16) ? 4 : 0);
      sAh[slot] = hi; sAl[slot] = lo;
    }
#pragma unroll
    for (int j = 0; j < 4; ++j) {
      int pairIdx = tid + 256 * j;
      int n = pairIdx >> 4, k = (pairIdx & 15) * 2;
      unsigned int hi, lo;
      split_pair(rw[j].x, rw[j].y, hi, lo);
      // B layout: lane = (n%16) + 16*((k>>3)&1); pair slot interleaves k blocks
      int slot = (((n >> 4) * 32 + (n & 15) + 16 * ((k >> 3) & 1)) << 3)
               + ((k & 7) >> 1) + ((k & 16) ? 4 : 0);
      sBh[slot] = hi; sBl[slot] = lo;
    }
  };
  auto compute = [&]() {
    v16bf ah[2], al[2], bh[2], bl[2];
#pragma unroll
    for (int i = 0; i < 2; ++i) {
      ah[i] = __builtin_bit_cast(v16bf, *(const v16us*)&sAh[((2 * wm + i) * 32 + lane) * 8]);
      al[i] = __builtin_bit_cast(v16bf, *(const v16us*)&sAl[((2 * wm + i) * 32 + lane) * 8]);
    }
#pragma unroll
    for (int j = 0; j < 2; ++j) {
      bh[j] = __builtin_bit_cast(v16bf, *(const v16us*)&sBh[((2 * wn + j) * 32 + lane) * 8]);
      bl[j] = __builtin_bit_cast(v16bf, *(const v16us*)&sBl[((2 * wn + j) * 32 + lane) * 8]);
    }
#pragma unroll
    for (int i = 0; i < 2; ++i)
#pragma unroll
      for (int j = 0; j < 2; ++j) {
        cc[i][j] = WMMA_BF16(ah[i], bh[j], cc[i][j]);   // hi*hi
        cc[i][j] = WMMA_BF16(ah[i], bl[j], cc[i][j]);   // hi*lo
        cc[i][j] = WMMA_BF16(al[i], bh[j], cc[i][j]);   // lo*hi
      }
  };

  // ---- software-pipelined main loop over the 192 full K tiles ----
  float2 rax[8], raw[4], rbx[8], rbw[4];
  loadX_fast(rax);                                   // tile 0
  if (fullN) loadW_fast(raw); else loadW_safe(raw, 0);
  for (int kt = 0; kt < NKT_FULL; kt += 2) {
    loadX_fast(rbx);                                 // tile kt+1 (always full-K)
    if (fullN) loadW_fast(rbw); else loadW_safe(rbw, kt + 1);
    storeT(rax, raw);
    __syncthreads();
    compute();
    __syncthreads();
    if (kt + 2 < NKT_FULL) {                         // tile kt+2
      loadX_fast(rax);
      if (fullN) loadW_fast(raw); else loadW_safe(raw, kt + 2);
    } else {                                         // tile 192: partial K
      loadX_safe(rax, kt + 2);
      loadW_safe(raw, kt + 2);
    }
    storeT(rbx, rbw);
    __syncthreads();
    compute();
    __syncthreads();
  }
  // ---- guarded tail: partial tile 192 (k = 6144..6155, zero-padded) ----
  storeT(rax, raw);
  __syncthreads();
  compute();

  // Fused epilogue: interactions = x*T, projected onto fin_w[:,O:], reduced
  // over the 16-lane N groups of the C layout, accumulated via LDS + atomics.
#pragma unroll
  for (int i = 0; i < 2; ++i) {
#pragma unroll
    for (int j = 0; j < 2; ++j) {
      int col = col0 + 32 * wn + 16 * j + lrow;
#pragma unroll
      for (int r = 0; r < 8; ++r) {
        int m = row0 + 32 * wm + 16 * i + r + 8 * hsel;
        float t = cc[i][j][r];
        float p0 = 0.0f, p1 = 0.0f, p2 = 0.0f;
        if (col < I_SZ) {
          float inter = x[(size_t)m * I_SZ + col] * t;
          p0 = inter * fin_w[0 * IO_SZ + O_SZ + col];
          p1 = inter * fin_w[1 * IO_SZ + O_SZ + col];
          p2 = inter * fin_w[2 * IO_SZ + O_SZ + col];
        }
#pragma unroll
        for (int off = 1; off < 16; off <<= 1) {
          p0 += __shfl_xor(p0, off, 32);
          p1 += __shfl_xor(p1, off, 32);
          p2 += __shfl_xor(p2, off, 32);
        }
        if (lrow == 0) {
          int lm = 32 * wm + 16 * i + r + 8 * hsel;
          atomicAdd(&sacc[lm * O_SZ + 0], p0);
          atomicAdd(&sacc[lm * O_SZ + 1], p1);
          atomicAdd(&sacc[lm * O_SZ + 2], p2);
        }
      }
    }
  }
  __syncthreads();
  for (int i = tid; i < BM * O_SZ; i += 256) {
    atomicAdd(&gacc[(size_t)(row0 + i / O_SZ) * O_SZ + (i % O_SZ)], sacc[i]);
  }
}

// ---------------------------------------------------------------------------
// Kernel 2: linear head + final combine.
// out[b,o] = sum_k (x[b,:]@lin_w[k,:] + lin_b[k]) * fin_w[o,k]
//          + gacc[b,o] + fin_b[o]
// ---------------------------------------------------------------------------
__global__ __launch_bounds__(256)
void finalize_kernel(const float* __restrict__ x,
                     const float* __restrict__ lin_w,
                     const float* __restrict__ lin_b,
                     const float* __restrict__ fin_w,
                     const float* __restrict__ fin_b,
                     const float* __restrict__ gacc,
                     float* __restrict__ out) {
  const int b = blockIdx.x;
  const float4* x4 = reinterpret_cast<const float4*>(x + (size_t)b * I_SZ);
  const float4* w0 = reinterpret_cast<const float4*>(lin_w + 0 * (size_t)I_SZ);
  const float4* w1 = reinterpret_cast<const float4*>(lin_w + 1 * (size_t)I_SZ);
  const float4* w2 = reinterpret_cast<const float4*>(lin_w + 2 * (size_t)I_SZ);

  float d0 = 0.0f, d1 = 0.0f, d2 = 0.0f;
  for (int j = threadIdx.x; j < I_SZ / 4; j += 256) {  // I_SZ % 4 == 0
    float4 xv = x4[j];
    float4 a = w0[j], bb = w1[j], cc = w2[j];
    d0 += xv.x * a.x + xv.y * a.y + xv.z * a.z + xv.w * a.w;
    d1 += xv.x * bb.x + xv.y * bb.y + xv.z * bb.z + xv.w * bb.w;
    d2 += xv.x * cc.x + xv.y * cc.y + xv.z * cc.z + xv.w * cc.w;
  }
  for (int off = 1; off < 32; off <<= 1) {
    d0 += __shfl_xor(d0, off, 32);
    d1 += __shfl_xor(d1, off, 32);
    d2 += __shfl_xor(d2, off, 32);
  }
  __shared__ float wr[8][3];
  int lane = threadIdx.x & 31, wid = threadIdx.x >> 5;
  if (lane == 0) { wr[wid][0] = d0; wr[wid][1] = d1; wr[wid][2] = d2; }
  __syncthreads();
  if (threadIdx.x == 0) {
    float s0 = 0.0f, s1 = 0.0f, s2 = 0.0f;
    for (int w = 0; w < 8; ++w) { s0 += wr[w][0]; s1 += wr[w][1]; s2 += wr[w][2]; }
    float ll0 = s0 + lin_b[0];
    float ll1 = s1 + lin_b[1];
    float ll2 = s2 + lin_b[2];
#pragma unroll
    for (int o = 0; o < O_SZ; ++o) {
      out[b * O_SZ + o] = ll0 * fin_w[o * IO_SZ + 0] +
                          ll1 * fin_w[o * IO_SZ + 1] +
                          ll2 * fin_w[o * IO_SZ + 2] +
                          gacc[b * O_SZ + o] + fin_b[o];
    }
  }
}

extern "C" void kernel_launch(void* const* d_in, const int* in_sizes, int n_in,
                              void* d_out, int out_size, void* d_ws, size_t ws_size,
                              hipStream_t stream) {
  const float* x     = (const float*)d_in[0];
  const float* lin_w = (const float*)d_in[1];
  const float* lin_b = (const float*)d_in[2];
  const float* w_int = (const float*)d_in[3];
  const float* fin_w = (const float*)d_in[4];
  const float* fin_b = (const float*)d_in[5];
  float* out  = (float*)d_out;
  float* gacc = (float*)d_ws;   // B*O fp32 accumulator (24 KB)

  const int n = B_SZ * O_SZ;
  zero_acc_kernel<<<(n + 255) / 256, 256, 0, stream>>>(gacc, n);

  dim3 grid((I_SZ + BN - 1) / BN, B_SZ / BM);   // 97 x 16 blocks
  interaction_gemm_kernel<<<grid, 256, 0, stream>>>(x, w_int, fin_w, gacc);

  finalize_kernel<<<B_SZ, 256, 0, stream>>>(x, lin_w, lin_b, fin_w, fin_b, gacc, out);
}
